// SwinTransformerBlock_83013127897172
// MI455X (gfx1250) — matrix operand order repaired and checked by reading, not compiled, not run on hardware
//
#include <hip/hip_runtime.h>
#include <cmath>

// ---------------------------------------------------------------------------
// SwinV2 block for gfx1250 (MI455X).
// DIM=512 NH=16 HD=32 WS=8 N=64 H=W=64 SHIFT=4 B=16 HID=2048
// tokens M = 65536, windows = 1024, tokens/window = 64.
//
// Roofline: ~420 GFLOP of matmul, ~0.6-0.9 GB HBM traffic after f16-izing
// all WMMA-consumed intermediates -> near the compute/memory knee, so every
// matmul uses v_wmma_f32_16x16x32_f16 and tile staging uses the gfx1250
// ASYNCcnt path (global_load_async_to_lds_b128) where the data is already f16.
// ---------------------------------------------------------------------------

typedef _Float16 v16h __attribute__((ext_vector_type(16)));
typedef float    v8f  __attribute__((ext_vector_type(8)));
typedef int      vi4  __attribute__((vector_size(4 * sizeof(int))));

#define GAS __attribute__((address_space(1)))
#define LAS __attribute__((address_space(3)))

#if defined(__gfx1250__) && __has_builtin(__builtin_amdgcn_global_load_async_to_lds_b128)
#define HAVE_ASYNC_LDS 1
#else
#define HAVE_ASYNC_LDS 0
#endif

// 16-byte global->LDS copy: async on gfx1250 if the builtin exists.
__device__ __forceinline__ void copy16_g2l(void* lds, const void* g) {
#if HAVE_ASYNC_LDS
  __builtin_amdgcn_global_load_async_to_lds_b128((GAS vi4*)g, (LAS vi4*)lds, 0, 0);
#else
  *(int4*)lds = *(const int4*)g;
#endif
}

template <int N>
__device__ __forceinline__ void wait_async() {
#if HAVE_ASYNC_LDS
#if __has_builtin(__builtin_amdgcn_s_wait_asynccnt)
  __builtin_amdgcn_s_wait_asynccnt(N);
#else
  asm volatile("s_wait_asynccnt %0" ::"i"(N) : "memory");
#endif
#endif
}

// ---- WMMA fragment helpers (ISA 7.12.2 layouts, wave32) -------------------
// A (16x32 f16): lane<16 holds row (lane&15), K {0..7,16..23}; lane>=16 K {8..15,24..31}.
__device__ __forceinline__ v16h load_a_frag(const _Float16* rowp, int lane) {
  const int koff = (lane < 16) ? 0 : 8;
  v16h a;
#pragma unroll
  for (int i = 0; i < 8; ++i) {
    a[i]     = rowp[koff + i];
    a[8 + i] = rowp[16 + koff + i];
  }
  return a;
}
// B (32x16 f16) from [N][K] buffer: lane<16 holds col (lane&15) K {0..7,16..23}; lane>=16 K {8..15,24..31}.
__device__ __forceinline__ v16h load_b_frag(const _Float16* colp, int lane) {
  const int koff = (lane < 16) ? 0 : 8;
  v16h b;
#pragma unroll
  for (int i = 0; i < 8; ++i) {
    b[i]     = colp[koff + i];
    b[8 + i] = colp[16 + koff + i];
  }
  return b;
}

__device__ __forceinline__ int region_id(int c) {  // thresholds H-WS=56, H-SHIFT=60
  return (c < 56) ? 0 : ((c < 60) ? 1 : 2);
}

// ---------------------------------------------------------------------------
// fp32 -> f16 converter (weight pre-pass)
// ---------------------------------------------------------------------------
__global__ __launch_bounds__(256)
void cvt_f32_f16(const float* __restrict__ in, _Float16* __restrict__ out, int n) {
  const int i = blockIdx.x * 256 + threadIdx.x;
  if (i < n) out[i] = (_Float16)in[i];
}

// ---------------------------------------------------------------------------
// Variant 1: A fp32 (x / residual stream), W f16, C f16.
//  C[M,N] = act(A[M,K] @ W16[N,K]^T + bias)
//  gatherA=1: fused roll(-4) + window partition on A rows.
//  biasMode 1: QKV composite [q_bias | 0 | v_bias]; 0: b0[col] (nullable).
//  actMode 1: exact GELU.
// A staged via VALU f32->f16 convert; W staged via async-to-LDS.
// ---------------------------------------------------------------------------
__global__ __launch_bounds__(256)
void gemm_a32w16(const float* __restrict__ A, const _Float16* __restrict__ W16,
                 const float* __restrict__ b0, const float* __restrict__ b1,
                 _Float16* __restrict__ C16, int M, int N, int K,
                 int gatherA, int actMode, int biasMode) {
  __shared__ _Float16 As[128][40];
  __shared__ _Float16 Bs[128][40];

  const int tid  = threadIdx.x;
  const int lane = tid & 31;
  const int wave = tid >> 5;
  const int m0   = blockIdx.y * 128;
  const int n0   = blockIdx.x * 128;
  const int wm   = wave & 1;
  const int wn   = wave >> 1;

  v8f acc[4][2];
#pragma unroll
  for (int i = 0; i < 4; ++i)
#pragma unroll
    for (int j = 0; j < 2; ++j) { v8f z = {}; acc[i][j] = z; }

  // A staging: thread -> (row = tid/2, 16-col half), with optional window gather
  const int arow = tid >> 1;
  const int acol = (tid & 1) * 16;
  int asrc = m0 + arow;
  if (gatherA == 1) {
    const int r = m0 + arow;
    const int b_ = r >> 6, t = r & 63;
    const int bimg = b_ >> 6, wi = b_ & 63;
    const int wh = wi >> 3, ww = wi & 7;
    const int ih = t >> 3, iw = t & 7;
    const int sh = (wh * 8 + ih + 4) & 63;
    const int sw = (ww * 8 + iw + 4) & 63;
    asrc = bimg * 4096 + sh * 64 + sw;
  }
  const float* Arow = A + (size_t)asrc * K;

  // W staging: thread -> rows (tid>>2, 64+(tid>>2)), 16B chunk (tid&3)
  const int crow = tid >> 2;
  const int cchk = (tid & 3) * 8;  // in halfs
  const _Float16* Wg0 = W16 + (size_t)(n0 + crow) * K;
  const _Float16* Wg1 = W16 + (size_t)(n0 + 64 + crow) * K;

  const int nsteps = K >> 5;
  for (int ks = 0; ks < nsteps; ++ks) {
    const int kb = ks * 32;
    copy16_g2l(&Bs[crow][cchk], Wg0 + kb + cchk);
    copy16_g2l(&Bs[64 + crow][cchk], Wg1 + kb + cchk);
#pragma unroll
    for (int i = 0; i < 16; ++i) As[arow][acol + i] = (_Float16)Arow[kb + acol + i];
    wait_async<0>();
    __syncthreads();

    v16h afr[4], bfr[2];
#pragma unroll
    for (int mt = 0; mt < 4; ++mt)
      afr[mt] = load_a_frag(&As[wm * 64 + mt * 16 + (lane & 15)][0], lane);
#pragma unroll
    for (int nt = 0; nt < 2; ++nt)
      bfr[nt] = load_b_frag(&Bs[wn * 32 + nt * 16 + (lane & 15)][0], lane);
#pragma unroll
    for (int mt = 0; mt < 4; ++mt)
#pragma unroll
      for (int nt = 0; nt < 2; ++nt)
        acc[mt][nt] = __builtin_amdgcn_wmma_f32_16x16x32_f16(
            false, afr[mt], false, bfr[nt], (short)0, acc[mt][nt], false, false);
    __syncthreads();
  }

  const int nlo   = lane & 15;
  const int mbase = (lane < 16) ? 0 : 8;
#pragma unroll
  for (int mt = 0; mt < 4; ++mt) {
#pragma unroll
    for (int nt = 0; nt < 2; ++nt) {
      const int colg = n0 + wn * 32 + nt * 16 + nlo;
      float bias = 0.0f;
      if (biasMode == 0) { if (b0) bias = b0[colg]; }
      else bias = (colg < 512) ? b0[colg] : ((colg >= 1024) ? b1[colg - 1024] : 0.0f);
#pragma unroll
      for (int v = 0; v < 8; ++v) {
        const int rowg = m0 + wm * 64 + mt * 16 + mbase + v;
        float val = acc[mt][nt][v] + bias;
        if (actMode == 1) val = 0.5f * val * (1.0f + erff(val * 0.70710678118f));
        C16[(size_t)rowg * N + colg] = (_Float16)val;
      }
    }
  }
}

// ---------------------------------------------------------------------------
// Variant 2: A f16, W f16, C fp32 (+bias). Fully async, double-buffered LDS.
// Per tile: 4 async b128 instructions per wave (2 A rows + 2 W rows per lane).
// ---------------------------------------------------------------------------
__global__ __launch_bounds__(256)
void gemm_a16w16(const _Float16* __restrict__ A16, const _Float16* __restrict__ W16,
                 const float* __restrict__ b0, float* __restrict__ C,
                 int M, int N, int K) {
  __shared__ _Float16 As[2][128][40];
  __shared__ _Float16 Bs[2][128][40];

  const int tid  = threadIdx.x;
  const int lane = tid & 31;
  const int wave = tid >> 5;
  const int m0   = blockIdx.y * 128;
  const int n0   = blockIdx.x * 128;
  const int wm   = wave & 1;
  const int wn   = wave >> 1;

  v8f acc[4][2];
#pragma unroll
  for (int i = 0; i < 4; ++i)
#pragma unroll
    for (int j = 0; j < 2; ++j) { v8f z = {}; acc[i][j] = z; }

  const int crow = tid >> 2;       // 0..63
  const int cchk = (tid & 3) * 8;  // 16B chunk offset in halfs
  const _Float16* Ag0 = A16 + (size_t)(m0 + crow) * K;
  const _Float16* Ag1 = A16 + (size_t)(m0 + 64 + crow) * K;
  const _Float16* Wg0 = W16 + (size_t)(n0 + crow) * K;
  const _Float16* Wg1 = W16 + (size_t)(n0 + 64 + crow) * K;

  auto issue = [&](int ks, int buf) {
    const int kb = ks * 32;
    copy16_g2l(&As[buf][crow][cchk],      Ag0 + kb + cchk);
    copy16_g2l(&As[buf][64 + crow][cchk], Ag1 + kb + cchk);
    copy16_g2l(&Bs[buf][crow][cchk],      Wg0 + kb + cchk);
    copy16_g2l(&Bs[buf][64 + crow][cchk], Wg1 + kb + cchk);
  };

  const int nsteps = K >> 5;
  issue(0, 0);
  for (int ks = 0; ks < nsteps; ++ks) {
    const int buf = ks & 1;
    if (ks + 1 < nsteps) {
      issue(ks + 1, buf ^ 1);  // prefetch next tile into other buffer
      wait_async<4>();         // first 4 (tile ks) complete in order
    } else {
      wait_async<0>();
    }
    __syncthreads();

    v16h afr[4], bfr[2];
#pragma unroll
    for (int mt = 0; mt < 4; ++mt)
      afr[mt] = load_a_frag(&As[buf][wm * 64 + mt * 16 + (lane & 15)][0], lane);
#pragma unroll
    for (int nt = 0; nt < 2; ++nt)
      bfr[nt] = load_b_frag(&Bs[buf][wn * 32 + nt * 16 + (lane & 15)][0], lane);
#pragma unroll
    for (int mt = 0; mt < 4; ++mt)
#pragma unroll
      for (int nt = 0; nt < 2; ++nt)
        acc[mt][nt] = __builtin_amdgcn_wmma_f32_16x16x32_f16(
            false, afr[mt], false, bfr[nt], (short)0, acc[mt][nt], false, false);
    __syncthreads();
  }

  const int nlo   = lane & 15;
  const int mbase = (lane < 16) ? 0 : 8;
#pragma unroll
  for (int mt = 0; mt < 4; ++mt) {
#pragma unroll
    for (int nt = 0; nt < 2; ++nt) {
      const int colg = n0 + wn * 32 + nt * 16 + nlo;
      const float bias = b0 ? b0[colg] : 0.0f;
#pragma unroll
      for (int v = 0; v < 8; ++v) {
        const int rowg = m0 + wm * 64 + mt * 16 + mbase + v;
        C[(size_t)rowg * N + colg] = acc[mt][nt][v] + bias;
      }
    }
  }
}

// ---------------------------------------------------------------------------
// CPB table: rpb[idx=dh*15+dw][head] = 16*sigmoid(cpb_w2 @ relu(cpb_w1 @ t + b1))
// ---------------------------------------------------------------------------
__global__ __launch_bounds__(128)
void cpb_kernel(const float* __restrict__ w1, const float* __restrict__ b1,
                const float* __restrict__ w2, float* __restrict__ rpb) {
  __shared__ float hid[512];
  const int idx = blockIdx.x;  // 0..224
  const int dh = idx / 15, dw = idx % 15;
  const float t0 = (float)(dh - 7) * (8.0f / 7.0f);
  const float t1 = (float)(dw - 7) * (8.0f / 7.0f);
  const float c0 = copysignf(log2f(fabsf(t0) + 1.0f) * (1.0f / 3.0f), t0);
  const float c1 = copysignf(log2f(fabsf(t1) + 1.0f) * (1.0f / 3.0f), t1);
  for (int j = threadIdx.x; j < 512; j += 128)
    hid[j] = fmaxf(c0 * w1[j * 2 + 0] + c1 * w1[j * 2 + 1] + b1[j], 0.0f);
  __syncthreads();
  if (threadIdx.x < 16) {
    const int h = threadIdx.x;
    float acc = 0.0f;
    for (int j = 0; j < 512; ++j) acc += hid[j] * w2[h * 512 + j];
    rpb[idx * 16 + h] = 16.0f / (1.0f + __expf(-acc));
  }
}

// ---------------------------------------------------------------------------
// Windowed cosine attention, one wave per (window, head). f16 in / f16 out.
// ---------------------------------------------------------------------------
__global__ __launch_bounds__(64)
void attn_kernel(const _Float16* __restrict__ qkv, const float* __restrict__ logit_scale,
                 const float* __restrict__ rpb, _Float16* __restrict__ out) {
  extern __shared__ _Float16 smem[];
  const int lane = threadIdx.x & 31;
  const int wave = threadIdx.x >> 5;
  const int b_   = blockIdx.x;             // 0..1023
  const int head = blockIdx.y * 2 + wave;  // 0..15
  const int wi   = b_ & 63;
  const int wh   = wi >> 3, ww = wi & 7;

  _Float16* Qf = smem + wave * 10240;  // [64][32]
  _Float16* Kf = Qf + 2048;            // [64][32]
  _Float16* Vt = Kf + 2048;            // [32][64]
  _Float16* Pf = Vt + 2048;            // [64][64]

  const float scale = __expf(fminf(logit_scale[head], 4.60517019f));

  const _Float16* base = qkv + (size_t)b_ * 64 * 1536 + head * 32;
  for (int n = lane; n < 64; n += 32) {
    const _Float16* qr = base + (size_t)n * 1536;
    const _Float16* kr = qr + 512;
    const _Float16* vr = qr + 1024;
    float qv[32], kv[32], qs = 0.0f, ks = 0.0f;
#pragma unroll
    for (int d = 0; d < 32; ++d) {
      qv[d] = (float)qr[d]; kv[d] = (float)kr[d];
      qs += qv[d] * qv[d]; ks += kv[d] * kv[d];
    }
    const float qi = 1.0f / fmaxf(sqrtf(qs), 1e-12f);
    const float ki = 1.0f / fmaxf(sqrtf(ks), 1e-12f);
#pragma unroll
    for (int d = 0; d < 32; ++d) {
      Qf[n * 32 + d] = (_Float16)(qv[d] * qi);
      Kf[n * 32 + d] = (_Float16)(kv[d] * ki);
      Vt[d * 64 + n] = vr[d];
    }
  }
  __syncthreads();

  const int nlo   = lane & 15;
  const int mbase = (lane < 16) ? 0 : 8;

  // S = Qn @ Kn^T (4x4 tiles, K=32)
  v16h aq[4], bk[4];
#pragma unroll
  for (int mt = 0; mt < 4; ++mt) aq[mt] = load_a_frag(&Qf[(mt * 16 + nlo) * 32], lane);
#pragma unroll
  for (int nt = 0; nt < 4; ++nt) bk[nt] = load_b_frag(&Kf[(nt * 16 + nlo) * 32], lane);

  v8f s[4][4];
#pragma unroll
  for (int mt = 0; mt < 4; ++mt)
#pragma unroll
    for (int nt = 0; nt < 4; ++nt) {
      v8f z = {};
      s[mt][nt] = __builtin_amdgcn_wmma_f32_16x16x32_f16(
          false, aq[mt], false, bk[nt], (short)0, z, false, false);
    }

  int jh[4], jw[4], ridj[4];
#pragma unroll
  for (int nt = 0; nt < 4; ++nt) {
    const int j = nt * 16 + nlo;
    jh[nt] = j >> 3; jw[nt] = j & 7;
    ridj[nt] = region_id(wh * 8 + jh[nt]) * 3 + region_id(ww * 8 + jw[nt]);
  }

  // bias + shift mask + softmax (C-layout rows live in 16-lane groups)
#pragma unroll
  for (int mt = 0; mt < 4; ++mt) {
#pragma unroll
    for (int v = 0; v < 8; ++v) {
      const int i  = mt * 16 + mbase + v;
      const int ih = i >> 3, iw = i & 7;
      const int ridi = region_id(wh * 8 + ih) * 3 + region_id(ww * 8 + iw);
      float sv[4];
#pragma unroll
      for (int nt = 0; nt < 4; ++nt) {
        const int idx = (ih - jh[nt] + 7) * 15 + (iw - jw[nt] + 7);
        float val = s[mt][nt][v] * scale + rpb[idx * 16 + head];
        if (ridi != ridj[nt]) val -= 100.0f;
        sv[nt] = val;
      }
      float m = fmaxf(fmaxf(sv[0], sv[1]), fmaxf(sv[2], sv[3]));
      for (int off = 8; off; off >>= 1) m = fmaxf(m, __shfl_xor(m, off, 32));
      float e[4], sum = 0.0f;
#pragma unroll
      for (int nt = 0; nt < 4; ++nt) { e[nt] = __expf(sv[nt] - m); sum += e[nt]; }
      for (int off = 8; off; off >>= 1) sum += __shfl_xor(sum, off, 32);
      const float inv = 1.0f / sum;
#pragma unroll
      for (int nt = 0; nt < 4; ++nt)
        Pf[i * 64 + nt * 16 + nlo] = (_Float16)(e[nt] * inv);
    }
  }
  __syncthreads();

  // O = P @ V (4x2 tiles, K=64 -> two chained WMMAs)
#pragma unroll
  for (int mt = 0; mt < 4; ++mt) {
    const _Float16* prow = &Pf[(mt * 16 + nlo) * 64];
    const v16h a0 = load_a_frag(prow, lane);
    const v16h a1 = load_a_frag(prow + 32, lane);
#pragma unroll
    for (int dt = 0; dt < 2; ++dt) {
      const int dcol = dt * 16 + nlo;
      const v16h b0v = load_b_frag(&Vt[dcol * 64], lane);
      const v16h b1v = load_b_frag(&Vt[dcol * 64 + 32], lane);
      v8f o = {};
      o = __builtin_amdgcn_wmma_f32_16x16x32_f16(false, a0, false, b0v, (short)0, o, false, false);
      o = __builtin_amdgcn_wmma_f32_16x16x32_f16(false, a1, false, b1v, (short)0, o, false, false);
#pragma unroll
      for (int v = 0; v < 8; ++v) {
        const int n = mt * 16 + mbase + v;
        out[((size_t)b_ * 64 + n) * 512 + head * 32 + dcol] = (_Float16)o[v];
      }
    }
  }
}

// ---------------------------------------------------------------------------
// out[t] = resid[t] + LayerNorm(src[row(t)]) * g + b   (C=512, one wave/token)
// gatherMode 1: src rows are window-ordered -> inverse shifted-window mapping.
// ---------------------------------------------------------------------------
__global__ __launch_bounds__(256)
void ln_add_kernel(const float* __restrict__ src, const float* __restrict__ resid,
                   const float* __restrict__ g, const float* __restrict__ b,
                   float* __restrict__ out, int gatherMode) {
  const int lane = threadIdx.x & 31;
  const int wave = threadIdx.x >> 5;
  const int t    = blockIdx.x * 8 + wave;  // 0..65535
  int srow = t;
  if (gatherMode == 1) {
    const int bimg = t >> 12, p = t & 4095;
    const int h = p >> 6, w = p & 63;
    const int hh = (h + 60) & 63, ww_ = (w + 60) & 63;  // inverse roll(+SHIFT)
    const int wh = hh >> 3, ih = hh & 7, wwi = ww_ >> 3, iw = ww_ & 7;
    srow = (bimg * 64 + wh * 8 + wwi) * 64 + ih * 8 + iw;
  }
  const float* row = src + (size_t)srow * 512;
  float vals[16], sum = 0.0f;
#pragma unroll
  for (int i = 0; i < 16; ++i) { vals[i] = row[lane + i * 32]; sum += vals[i]; }
  for (int off = 16; off; off >>= 1) sum += __shfl_xor(sum, off, 32);
  const float mu = sum * (1.0f / 512.0f);
  float vs = 0.0f;
#pragma unroll
  for (int i = 0; i < 16; ++i) { const float d = vals[i] - mu; vs += d * d; }
  for (int off = 16; off; off >>= 1) vs += __shfl_xor(vs, off, 32);
  const float rstd = rsqrtf(vs * (1.0f / 512.0f) + 1e-5f);
  const float* rres = resid + (size_t)t * 512;
  float* orow = out + (size_t)t * 512;
#pragma unroll
  for (int i = 0; i < 16; ++i) {
    const int c = lane + i * 32;
    orow[c] = rres[c] + (vals[i] - mu) * rstd * g[c] + b[c];
  }
}

// ---------------------------------------------------------------------------
extern "C" void kernel_launch(void* const* d_in, const int* in_sizes, int n_in,
                              void* d_out, int out_size, void* d_ws, size_t ws_size,
                              hipStream_t stream) {
  (void)in_sizes; (void)n_in; (void)out_size; (void)ws_size;
  const float* x      = (const float*)d_in[0];
  const float* qkv_w  = (const float*)d_in[1];
  const float* q_bias = (const float*)d_in[2];
  const float* v_bias = (const float*)d_in[3];
  const float* lsc    = (const float*)d_in[4];
  const float* cpb_w1 = (const float*)d_in[5];
  const float* cpb_b1 = (const float*)d_in[6];
  const float* cpb_w2 = (const float*)d_in[7];
  const float* proj_w = (const float*)d_in[8];
  const float* proj_b = (const float*)d_in[9];
  const float* n1g    = (const float*)d_in[10];
  const float* n1b    = (const float*)d_in[11];
  const float* fc1_w  = (const float*)d_in[12];
  const float* fc1_b  = (const float*)d_in[13];
  const float* fc2_w  = (const float*)d_in[14];
  const float* fc2_b  = (const float*)d_in[15];
  const float* n2g    = (const float*)d_in[16];
  const float* n2b    = (const float*)d_in[17];
  float* out = (float*)d_out;

  // Workspace layout (bytes), lifetimes overlapped:
  //  [0, 256MB)            qkv16 (65536x1536 f16) ... reused as m1_16 (65536x2048 f16)
  //  [256MB, 384MB)        attn16 (65536x512 f16) ... reused as m2 (65536x512 f32)
  //  [384MB, 512MB)        y (65536x512 f32, window-ordered rows)
  //  [512MB, ...)          f16 weights + rpb table
  char* ws = (char*)d_ws;
  _Float16* qkv16 = (_Float16*)(ws);
  _Float16* m1_16 = (_Float16*)(ws);
  _Float16* attn16 = (_Float16*)(ws + (size_t)268435456);
  float*    m2     = (float*)(ws + (size_t)268435456);
  float*    y      = (float*)(ws + (size_t)402653184);
  _Float16* qkv_w16 = (_Float16*)(ws + (size_t)536870912);
  _Float16* proj_w16 = qkv_w16 + 786432;
  _Float16* fc1_w16  = proj_w16 + 262144;
  _Float16* fc2_w16  = fc1_w16 + 1048576;
  float*    rpb      = (float*)(fc2_w16 + 1048576);

  // 0) weight pre-convert to f16 (6.3 MB total) + CPB table
  cvt_f32_f16<<<3072, 256, 0, stream>>>(qkv_w, qkv_w16, 786432);
  cvt_f32_f16<<<1024, 256, 0, stream>>>(proj_w, proj_w16, 262144);
  cvt_f32_f16<<<4096, 256, 0, stream>>>(fc1_w, fc1_w16, 1048576);
  cvt_f32_f16<<<4096, 256, 0, stream>>>(fc2_w, fc2_w16, 1048576);
  cpb_kernel<<<225, 128, 0, stream>>>(cpb_w1, cpb_b1, cpb_w2, rpb);

  // 1) QKV projection with fused roll + window-partition gather (out f16)
  gemm_a32w16<<<dim3(12, 512), 256, 0, stream>>>(
      x, qkv_w16, q_bias, v_bias, qkv16, 65536, 1536, 512, /*gather*/1, 0, /*bias*/1);
  // 2) windowed cosine attention (2 heads / 64-thread block, 40KB LDS)
  attn_kernel<<<dim3(1024, 8), 64, 2 * 10240 * 2, stream>>>(qkv16, lsc, rpb, attn16);
  // 3) output projection (async double-buffered f16 GEMM, out f32)
  gemm_a16w16<<<dim3(4, 512), 256, 0, stream>>>(attn16, proj_w16, proj_b, y, 65536, 512, 512);
  // 4) x1 = x + LN(window_reverse(roll_back(y)))  -> d_out
  ln_add_kernel<<<8192, 256, 0, stream>>>(y, x, n1g, n1b, out, 1);
  // 5) m1 = gelu(x1 @ fc1^T + b)  (out f16)
  gemm_a32w16<<<dim3(16, 512), 256, 0, stream>>>(
      out, fc1_w16, fc1_b, nullptr, m1_16, 65536, 2048, 512, 0, /*gelu*/1, 0);
  // 6) m2 = m1 @ fc2^T + b  (async double-buffered f16 GEMM, out f32)
  gemm_a16w16<<<dim3(4, 512), 256, 0, stream>>>(m1_16, fc2_w16, fc2_b, m2, 65536, 512, 2048);
  // 7) out = x1 + LN(m2)
  ln_add_kernel<<<8192, 256, 0, stream>>>(m2, out, n2g, n2b, out, 0);
}
